// Encoder_76493367542408
// MI455X (gfx1250) — compile-verified, matrix operand order checked by reading, not run on hardware
//
#include <hip/hip_runtime.h>
#include <hip/hip_bf16.h>
#include <math.h>

typedef __attribute__((ext_vector_type(16))) _Float16 v16h;
typedef __attribute__((ext_vector_type(8)))  _Float16 v8h;
typedef __attribute__((ext_vector_type(8)))  float    v8f;
typedef __attribute__((ext_vector_type(4)))  unsigned int v4u;
typedef __attribute__((ext_vector_type(4)))  int      v4i;
typedef __attribute__((ext_vector_type(8)))  int      v8i;

#define WMMA_F16(a,b,c) __builtin_amdgcn_wmma_f32_16x16x32_f16(false,(a),false,(b),(short)0,(c),false,false)

// Build a v16h A/B fragment per the CDNA5 16-bit WMMA layout:
// lane holds K = {khalf..khalf+7} in elems 0..7 and K = {khalf+16..khalf+23} in elems 8..15.
template<typename T>
static __device__ inline v16h ldfrag(const T* p) {
  v8h lo = *(const v8h*)(p);
  v8h hi = *(const v8h*)(p + 16);
  v16h r;
#pragma unroll
  for (int i = 0; i < 8; ++i) { r[i] = lo[i]; r[8 + i] = hi[i]; }
  return r;
}

// ---------------------------------------------------------------------------
// TDM: DMA a 2D tile (rows x tile_k f16 elements, row stride = stride_elems)
// from global memory into LDS at lds_addr (contiguous [rows][tile_k]).
// Descriptor layout per CDNA5 ISA ch.8 (D# groups 0/1); groups 2/3 zero (2D).
// Issue from ONE wave only; completion via TENSORcnt.
// ---------------------------------------------------------------------------
static __device__ inline void tdm_load_2d(unsigned lds_addr, const void* gptr,
                                          unsigned tile_k, unsigned rows,
                                          unsigned long long stride_elems)
{
  const unsigned long long ga = (unsigned long long)(size_t)gptr;
  v4u g0;
  g0[0] = 1u;                                            // count=1, user D#
  g0[1] = lds_addr;                                      // LDS byte address
  g0[2] = (unsigned)(ga & 0xffffffffull);                // global_addr[31:0]
  g0[3] = (unsigned)((ga >> 32) & 0x1ffffffull)          // global_addr[56:32]
        | (2u << 30);                                    // type = 2 ("image")
  const unsigned td0 = 0x40000000u, td1 = 0x40000000u;   // huge dims: no OOB clamp
  v8i g1;
  g1[0] = (int)(1u << 16);                               // data_size = 1 -> 2 bytes
  g1[1] = (int)((td0 & 0xffffu) << 16);                  // tensor_dim0[15:0]
  g1[2] = (int)(((td0 >> 16) & 0xffffu) | ((td1 & 0xffffu) << 16));
  g1[3] = (int)(((td1 >> 16) & 0xffffu) | (tile_k << 16)); // tile_dim0
  g1[4] = (int)(rows & 0xffffu);                         // tile_dim1 (tile_dim2=0)
  g1[5] = (int)(unsigned)(stride_elems & 0xffffffffull); // dim0 stride[31:0]
  g1[6] = (int)(unsigned)((stride_elems >> 32) & 0xffffull);
  g1[7] = 0;
  const v4i z4 = {0, 0, 0, 0};
#if __has_include(<hip/amd_detail/amd_gfx1250_TDM.h>)
  const v8i z8 = {0, 0, 0, 0, 0, 0, 0, 0};
  __builtin_amdgcn_tensor_load_to_lds(g0, g1, z4, z4, z8, 0);
#else
  __builtin_amdgcn_tensor_load_to_lds(g0, g1, z4, z4, 0);
#endif
}

// ---------------------------------------------------------------------------
// WMMA GEMM: C[M,N] = A[M,K](f16 row-major) * BT[N,K](f16) + bias
// block = 256 threads = 8 waves; wave = 16x32 tile; workgroup = 64x64 tile.
// A/B slabs staged into LDS by the Tensor Data Mover, K chunked by 256.
// EPI: 1 = QKV remap -> f16   2 = exact GELU -> f16   3 = +residual -> f32
// ---------------------------------------------------------------------------
template<int EPI>
__global__ void __launch_bounds__(256)
gemm_wmma(const _Float16* __restrict__ A, const _Float16* __restrict__ BT,
          const float* __restrict__ bias, const float* __restrict__ resid,
          float* __restrict__ outF, _Float16* __restrict__ outH,
          int M, int N, int K)
{
  __shared__ __align__(16) _Float16 As[64 * 256];
  __shared__ __align__(16) _Float16 Bs[64 * 256];

  const int lane = threadIdx.x & 31;
  const int wid  = threadIdx.x >> 5;
  const int wm = wid & 3, wn = wid >> 2;
  const int row0 = blockIdx.y * 64;
  const int col0 = blockIdx.x * 64;
  const int r  = lane & 15;
  const int hi = lane >> 4;
  const int khalf = hi << 3;

  v8f c0 = {}; v8f c1 = {};
  for (int kc = 0; kc < K; kc += 256) {
    if (kc) __syncthreads();                // LDS reuse: previous chunk fully read
    if (wid == 0) {
      tdm_load_2d((unsigned)(size_t)As, A  + (size_t)row0 * K + kc, 256u, 64u,
                  (unsigned long long)K);
      tdm_load_2d((unsigned)(size_t)Bs, BT + (size_t)col0 * K + kc, 256u, 64u,
                  (unsigned long long)K);
      __builtin_amdgcn_s_wait_tensorcnt(0);
    }
    __syncthreads();

    const _Float16* pa  = &As[(wm * 16 + r) * 256 + khalf];
    const _Float16* pb0 = &Bs[(wn * 32 + r) * 256 + khalf];
    const _Float16* pb1 = &Bs[(wn * 32 + 16 + r) * 256 + khalf];
#pragma unroll
    for (int kk = 0; kk < 256; kk += 32) {
      v16h a  = ldfrag(pa  + kk);
      v16h b0 = ldfrag(pb0 + kk);
      v16h b1 = ldfrag(pb1 + kk);
      c0 = WMMA_F16(a, b0, c0);
      c1 = WMMA_F16(a, b1, c1);
    }
  }

  const int cA = col0 + wn * 32 + r;
  const int cB = cA + 16;
  const float biasA = bias[cA], biasB = bias[cB];
#pragma unroll
  for (int i = 0; i < 8; ++i) {
    const int row = row0 + wm * 16 + i + (hi << 3);   // C layout: lanes 16-31 -> +8
    float v0 = c0[i] + biasA;
    float v1 = c1[i] + biasB;
    if (EPI == 3) {
      const size_t o0 = (size_t)row * N + cA;
      const size_t o1 = (size_t)row * N + cB;
      outF[o0] = v0 + resid[o0];
      outF[o1] = v1 + resid[o1];
    } else if (EPI == 2) {
      v0 = 0.5f * v0 * (1.0f + erff(v0 * 0.70710678118654752f));
      v1 = 0.5f * v1 * (1.0f + erff(v1 * 0.70710678118654752f));
      outH[(size_t)row * N + cA] = (_Float16)v0;
      outH[(size_t)row * N + cB] = (_Float16)v1;
    } else { // EPI == 1: scatter into QKV[b][sect*8+h][t][e], sect = col/256
      const int b = row >> 7, t = row & 127;
      outH[(((size_t)b * 24 + (cA >> 5)) * 128 + t) * 32 + (cA & 31)] = (_Float16)v0;
      outH[(((size_t)b * 24 + (cB >> 5)) * 128 + t) * 32 + (cB & 31)] = (_Float16)v1;
    }
  }
}

// ---------------------------------------------------------------------------
// Fused attention for one (batch, head): S = QK^T*scale + mask, softmax, P*V.
// 256 threads = 8 waves; wave w owns query rows [16w,16w+16).
// K tile staged by TDM; V staged transposed for contiguous PV B-fragments.
// ---------------------------------------------------------------------------
__global__ void __launch_bounds__(256)
attn_kernel(const _Float16* __restrict__ QKV, const float* __restrict__ neg,
            _Float16* __restrict__ Z)
{
  const int bh = blockIdx.x;
  const int b = bh >> 3, h = bh & 7;
  const _Float16* Qp = QKV + (size_t)(b * 24 + h)      * 4096;
  const _Float16* Kp = QKV + (size_t)(b * 24 + 8 + h)  * 4096;
  const _Float16* Vp = QKV + (size_t)(b * 24 + 16 + h) * 4096;

  __shared__ __align__(16) _Float16 Ks[128 * 32];   // K row-major [t][e]
  __shared__ __align__(16) _Float16 Vt[32 * 128];   // V transposed [e][t]
  __shared__ __align__(16) _Float16 P[8][16 * 128]; // per-wave probs

  const int tid = threadIdx.x;
  const int w = tid >> 5, lane = tid & 31;
  if (w == 0) {
    tdm_load_2d((unsigned)(size_t)Ks, Kp, 4096u, 1u, 4096ull);
    __builtin_amdgcn_s_wait_tensorcnt(0);
  }
  for (int i = tid; i < 4096; i += 256) {
    int t = i >> 5, e = i & 31;
    Vt[e * 128 + t] = Vp[i];
  }
  __syncthreads();

  const int r = lane & 15, hi = lane >> 4, khalf = hi << 3;
  const int m0 = w * 16;
  const float scale = 0.17677669529663687f; // 1/sqrt(32)

  const v16h qa = ldfrag(Qp + (size_t)(m0 + r) * 32 + khalf);

  v8f sc[8];
#pragma unroll
  for (int nt = 0; nt < 8; ++nt) {
    v16h kb = ldfrag(&Ks[(nt * 16 + r) * 32 + khalf]);
    v8f acc = {};
    acc = WMMA_F16(qa, kb, acc);
    const float ng = neg[b * 128 + nt * 16 + r]; // per-key mask bias
#pragma unroll
    for (int i = 0; i < 8; ++i) sc[nt][i] = acc[i] * scale + ng;
  }

  // Row-wise softmax in registers: row = m0 + i + 8*hi spans 16 lanes x 8 tiles.
#pragma unroll
  for (int i = 0; i < 8; ++i) {
    float mx = -1e30f;
#pragma unroll
    for (int nt = 0; nt < 8; ++nt) mx = fmaxf(mx, sc[nt][i]);
    mx = fmaxf(mx, __shfl_xor(mx, 1, 32));
    mx = fmaxf(mx, __shfl_xor(mx, 2, 32));
    mx = fmaxf(mx, __shfl_xor(mx, 4, 32));
    mx = fmaxf(mx, __shfl_xor(mx, 8, 32));
    float sum = 0.f;
#pragma unroll
    for (int nt = 0; nt < 8; ++nt) { float e = __expf(sc[nt][i] - mx); sc[nt][i] = e; sum += e; }
    sum += __shfl_xor(sum, 1, 32);
    sum += __shfl_xor(sum, 2, 32);
    sum += __shfl_xor(sum, 4, 32);
    sum += __shfl_xor(sum, 8, 32);
    const float inv = 1.0f / sum;
    const int prow = i + (hi << 3);
#pragma unroll
    for (int nt = 0; nt < 8; ++nt)
      P[w][prow * 128 + nt * 16 + r] = (_Float16)(sc[nt][i] * inv);
  }
  __syncthreads();

  // z[16,32] = P[16,128] * V[128,32]
#pragma unroll
  for (int et = 0; et < 2; ++et) {
    v8f z = {};
#pragma unroll
    for (int kk = 0; kk < 128; kk += 32) {
      v16h pa = ldfrag(&P[w][r * 128 + kk + khalf]);
      v16h vb = ldfrag(&Vt[(et * 16 + r) * 128 + kk + khalf]);
      z = WMMA_F16(pa, vb, z);
    }
#pragma unroll
    for (int i = 0; i < 8; ++i) {
      const int trow = m0 + i + (hi << 3);
      Z[((size_t)(b * 128 + trow)) * 256 + h * 32 + et * 16 + r] = (_Float16)z[i];
    }
  }
}

// ---------------------------------------------------------------------------
// LayerNorm: one wave per 256-wide row; writes fp32 + f16 copies.
// ---------------------------------------------------------------------------
__global__ void __launch_bounds__(256)
ln_kernel(const float* __restrict__ in, const float* __restrict__ g,
          const float* __restrict__ bb, float* __restrict__ outF,
          _Float16* __restrict__ outH)
{
  const int w = threadIdx.x >> 5, lane = threadIdx.x & 31;
  const int row = blockIdx.x * 8 + w;
  const float* p = in + (size_t)row * 256;
  float v[8]; float s = 0.f;
#pragma unroll
  for (int i = 0; i < 8; ++i) { v[i] = p[lane + 32 * i]; s += v[i]; }
#pragma unroll
  for (int m = 1; m < 32; m <<= 1) s += __shfl_xor(s, m, 32);
  const float mu = s * (1.0f / 256.0f);
  float var = 0.f;
#pragma unroll
  for (int i = 0; i < 8; ++i) { float d = v[i] - mu; var += d * d; }
#pragma unroll
  for (int m = 1; m < 32; m <<= 1) var += __shfl_xor(var, m, 32);
  const float inv = rsqrtf(var * (1.0f / 256.0f) + 1e-6f);
#pragma unroll
  for (int i = 0; i < 8; ++i) {
    const int c = lane + 32 * i;
    const float o = (v[i] - mu) * inv * g[c] + bb[c];
    outF[(size_t)row * 256 + c] = o;
    outH[(size_t)row * 256 + c] = (_Float16)o;
  }
}

// ---------------------------------------------------------------------------
// Small prep kernels
// ---------------------------------------------------------------------------
__global__ void mask_kernel(const float* __restrict__ x_inp, float* __restrict__ neg)
{
  const int i = blockIdx.x * blockDim.x + threadIdx.x; // b*128 + t
  const float* p = x_inp + (size_t)i * 164;
  float s = 0.f;
  for (int j = 0; j < 164; ++j) s += p[j];
  neg[i] = (s == 0.0f) ? -1e9f : 0.0f;
}

__global__ void cvt_in(const float* __restrict__ x, float* __restrict__ Xf,
                       _Float16* __restrict__ Xh)
{
  const int i = blockIdx.x * blockDim.x + threadIdx.x;
  const float v = x[i];
  Xf[i] = v;
  Xh[i] = (_Float16)v;
}

// dst[n*K+k] = src[k*N+n]  (row-major [K][N] -> BT [N][K], fp32 -> f16)
__global__ void pack_rm(const float* __restrict__ src, _Float16* __restrict__ dst,
                        int K, int N)
{
  const int i = blockIdx.x * blockDim.x + threadIdx.x; // over N*K
  const int n = i / K, k = i - n * K;
  dst[i] = (_Float16)src[(size_t)k * N + n];
}

// wq/wk/wv [H=8][K=256][E=32] -> BT [n=h*32+e][k], K fixed = 256
__global__ void pack_head(const float* __restrict__ src, _Float16* __restrict__ dst)
{
  const int i = blockIdx.x * blockDim.x + threadIdx.x; // over 256*256
  const int n = i >> 8, k = i & 255;
  dst[i] = (_Float16)src[(((size_t)(n >> 5) * 256 + k) << 5) + (n & 31)];
}

__global__ void bias3(const float* __restrict__ bq, const float* __restrict__ bk,
                      const float* __restrict__ bv, float* __restrict__ dst)
{
  const int i = blockIdx.x * blockDim.x + threadIdx.x; // 0..767
  dst[i] = (i < 256) ? bq[i] : (i < 512) ? bk[i - 256] : bv[i - 512];
}

// ---------------------------------------------------------------------------
extern "C" void kernel_launch(void* const* d_in, const int* in_sizes, int n_in,
                              void* d_out, int out_size, void* d_ws, size_t ws_size,
                              hipStream_t stream)
{
  (void)in_sizes; (void)n_in; (void)out_size; (void)ws_size;
  const int M = 128 * 128;      // tokens
  const int D = 256, FF = 1024;

  const float* x    = (const float*)d_in[0];
  const float* xinp = (const float*)d_in[1];

  char* ws = (char*)d_ws; size_t off = 0;
  auto alloc = [&](size_t bytes) -> void* {
    void* p = ws + off; off += (bytes + 255) & ~(size_t)255; return p;
  };
  float*    negm = (float*)   alloc((size_t)M * 4);
  float*    Xcur = (float*)   alloc((size_t)M * D * 4);
  _Float16* X16  = (_Float16*)alloc((size_t)M * D * 2);
  _Float16* Wqkv = (_Float16*)alloc((size_t)768 * 256 * 2);
  float*    bqkv = (float*)   alloc((size_t)768 * 4);
  _Float16* WoT  = (_Float16*)alloc((size_t)256 * 256 * 2);
  _Float16* W1T  = (_Float16*)alloc((size_t)1024 * 256 * 2);
  _Float16* W2T  = (_Float16*)alloc((size_t)256 * 1024 * 2);
  _Float16* QKV  = (_Float16*)alloc((size_t)128 * 24 * 128 * 32 * 2);
  _Float16* Zb   = (_Float16*)alloc((size_t)M * D * 2);
  float*    Tmp  = (float*)   alloc((size_t)M * D * 4);
  _Float16* Hb   = (_Float16*)alloc((size_t)M * FF * 2);

  mask_kernel<<<M / 256, 256, 0, stream>>>(xinp, negm);
  cvt_in<<<(M * D) / 256, 256, 0, stream>>>(x, Xcur, X16);

  for (int p = 0; p < 3; ++p) {
    // params in dict-insertion order:
    // 0 wq 1 bq 2 wk 3 bk 4 wv 5 bv 6 wo 7 bo 8 ln1_g 9 ln1_b
    // 10 w1 11 b1 12 w2 13 b2 14 ln2_g 15 ln2_b
    const float* pp[16];
    for (int j = 0; j < 16; ++j) pp[j] = (const float*)d_in[2 + p * 16 + j];

    pack_head<<<256, 256, 0, stream>>>(pp[0], Wqkv);
    pack_head<<<256, 256, 0, stream>>>(pp[2], Wqkv + 256 * 256);
    pack_head<<<256, 256, 0, stream>>>(pp[4], Wqkv + 512 * 256);
    bias3<<<3, 256, 0, stream>>>(pp[1], pp[3], pp[5], bqkv);
    pack_rm<<<(256 * 256) / 256, 256, 0, stream>>>(pp[6], WoT, 256, 256);
    pack_rm<<<(256 * 1024) / 256, 256, 0, stream>>>(pp[10], W1T, 256, 1024);
    pack_rm<<<(1024 * 256) / 256, 256, 0, stream>>>(pp[12], W2T, 1024, 256);

    // fused QKV projection -> QKV[b][24][t][32]
    gemm_wmma<1><<<dim3(768 / 64, M / 64), 256, 0, stream>>>(
        X16, Wqkv, bqkv, nullptr, nullptr, QKV, M, 768, 256);

    // attention per (b,h)
    attn_kernel<<<128 * 8, 256, 0, stream>>>(QKV, negm, Zb);

    // output projection + bias + residual -> Tmp (fp32)
    gemm_wmma<3><<<dim3(256 / 64, M / 64), 256, 0, stream>>>(
        Zb, WoT, pp[7], Xcur, Tmp, nullptr, M, 256, 256);
    ln_kernel<<<M / 8, 256, 0, stream>>>(Tmp, pp[8], pp[9], Xcur, X16);

    // FFN
    gemm_wmma<2><<<dim3(1024 / 64, M / 64), 256, 0, stream>>>(
        X16, W1T, pp[11], nullptr, nullptr, Hb, M, 1024, 256);
    gemm_wmma<3><<<dim3(256 / 64, M / 64), 256, 0, stream>>>(
        Hb, W2T, pp[13], Xcur, Tmp, nullptr, M, 256, 1024);
    float* lnout = (p == 2) ? (float*)d_out : Xcur;
    ln_kernel<<<M / 8, 256, 0, stream>>>(Tmp, pp[14], pp[15], lnout, X16);
  }
}
